// SnakeSimpleLoss_53618371723453
// MI455X (gfx1250) — matrix operand order repaired and checked by reading, not compiled, not run on hardware
//
#include <hip/hip_runtime.h>
#include <math.h>
#include <stdint.h>

typedef _Float16 half_t;
typedef __attribute__((ext_vector_type(16))) _Float16 v16h;
typedef __attribute__((ext_vector_type(8)))  float    v8f;

#define TILE_W   64
#define TILE_H   32
#define IN_W     72          // TILE_W + 8 halo (= 18 float4 chunks per row)
#define IN_H     40          // TILE_H + 8 halo
#define IN_PITCH 76          // floats; multiple of 4 -> 16B-aligned b128 chunks
#define H_PITCH  48          // 40 rows padded (room for K=24..31 guard reads)

struct Filt { float g[9]; float dg[9]; };

__device__ __forceinline__ v16h load_B(const half_t* __restrict__ Hl,
                                       int col, int ly0, int hi) {
    // 16-bit B 32x16 layout: VGPR v<4, lanes 0-15 hold K=2v,2v+1, lanes 16-31 K=8+2v,9+2v;
    // VGPR v>=4 holds K=16+2(v-4)(+8 for lanes 16-31).  Live K = 0..23 (9-tap band), K>=24 -> 0.
    v16h bb;
    const half_t* base = Hl + col * H_PITCH + ly0;
#pragma unroll
    for (int v = 0; v < 8; ++v) {
        int k0 = (v < 4) ? (2 * v + hi * 8) : (16 + 2 * (v - 4) + hi * 8);
        int kk0 = (k0     < 24) ? k0     : 0;
        int kk1 = (k0 + 1 < 24) ? k0 + 1 : 0;
        half_t b0 = base[kk0];
        half_t b1 = base[kk1];
        bb[2 * v]     = (k0     < 24) ? b0 : (half_t)0.f;
        bb[2 * v + 1] = (k0 + 1 < 24) ? b1 : (half_t)0.f;
    }
    return bb;
}

__global__ __launch_bounds__(256)
void snake_conv_loss_kernel(const float* __restrict__ pred,
                            const float* __restrict__ snake,
                            float* __restrict__ gimg,     // [16,2,1024,1024]
                            float* __restrict__ gimgW,    // [16,2,1024,1024]
                            float* __restrict__ partials,
                            Filt f)
{
    __shared__ float  sIn[IN_H * IN_PITCH];        // pred tile + halo
    __shared__ half_t sHg  [TILE_W * H_PITCH];     // horiz g   of pred   (col-major)
    __shared__ half_t sHdg [TILE_W * H_PITCH];     // horiz dg  of pred
    __shared__ half_t sAHg [TILE_W * H_PITCH];     // horiz g   of |pred|
    __shared__ half_t sAHdg[TILE_W * H_PITCH];     // horiz dg  of |pred|
    __shared__ float  sW[32];                      // [0..8]=g, [16..24]=dg
    __shared__ float  sRed[256];

    const int t  = threadIdx.x;
    const int b  = blockIdx.z;
    const int y0 = blockIdx.y * TILE_H;
    const int x0 = blockIdx.x * TILE_W;

    if (t == 0) {
#pragma unroll
        for (int i = 0; i < 9; ++i) { sW[i] = f.g[i]; sW[16 + i] = f.dg[i]; }
    }

    const float* predB  = pred  + (size_t)b * 1024 * 1024;
    const float* snakeB = snake + (size_t)b * 1024 * 1024;

    // ---- Phase 1: async-stage pred tile (+ zero halo) straight into LDS ----
    // 18 b128 chunks per row x 40 rows = 720 chunks; halo width = exactly one
    // chunk, x0 is 64-aligned, so each chunk is fully in- or out-of-bounds.
    for (int i = t; i < IN_H * (IN_W / 4); i += 256) {
        int r  = i / 18;               // tile row (0..39)
        int c4 = i - r * 18;           // float4 chunk within row (0..17)
        int gy = y0 - 4 + r;
        int gx = x0 - 4 + c4 * 4;
        unsigned lds_off = (unsigned)(uintptr_t)&sIn[r * IN_PITCH + c4 * 4];
        bool inb = (gy >= 0) & (gy < 1024) & (gx >= 0) & (gx < 1024);
        if (inb) {
            const float* gp = predB + (size_t)gy * 1024 + gx;
            asm volatile("global_load_async_to_lds_b128 %0, %1, off"
                         :: "v"(lds_off), "v"(gp) : "memory");
        } else {
            float4 z = make_float4(0.f, 0.f, 0.f, 0.f);
            *(float4*)&sIn[r * IN_PITCH + c4 * 4] = z;
        }
    }
    asm volatile("s_wait_asynccnt 0" ::: "memory");
    __syncthreads();

    // ---- Phase 2: horizontal separable passes (VALU), store f16 col-major ----
    for (int i = t; i < TILE_W * IN_H; i += 256) {
        int col = i / IN_H, row = i - col * IN_H;
        const float* ip = &sIn[row * IN_PITCH + col];
        float hg = 0.f, hdg = 0.f, ahg = 0.f, ahdg = 0.f;
#pragma unroll
        for (int j = 0; j < 9; ++j) {
            float v  = ip[j];
            float av = fabsf(v);
            hg   = fmaf(f.g[j],  v,  hg);
            hdg  = fmaf(f.dg[j], v,  hdg);
            ahg  = fmaf(f.g[j],  av, ahg);
            ahdg = fmaf(f.dg[j], av, ahdg);
        }
        int hidx = col * H_PITCH + row;
        sHg[hidx]   = (half_t)hg;
        sHdg[hidx]  = (half_t)hdg;
        sAHg[hidx]  = (half_t)ahg;
        sAHdg[hidx] = (half_t)ahdg;
    }
    __syncthreads();

    // ---- Phase 3: vertical passes as banded-weight WMMA, one 16x16 tile/wave ----
    const int lane = t & 31;
    const int wv   = t >> 5;
    const int tc   = (wv & 3) * 16;     // tile col offset within 64
    const int ly0  = (wv >> 2) * 16;    // tile row offset within 32
    const int m    = lane & 15;
    const int hi   = lane >> 4;

    // A (16x32 f16): A[m][k] = w[k-m] for 0<=k-m<=8 (tap j <-> offset j-4)
    v16h a_dy, a_dx;
#pragma unroll
    for (int v = 0; v < 8; ++v) {
        int k0 = (v < 4) ? (2 * v + hi * 8) : (16 + 2 * (v - 4) + hi * 8);
        int t0 = k0 - m, t1 = k0 + 1 - m;
        int c0 = min(max(t0, 0), 8), c1 = min(max(t1, 0), 8);
        bool v0 = (t0 >= 0) & (t0 < 9), v1 = (t1 >= 0) & (t1 < 9);
        float dg0 = sW[16 + c0], dg1 = sW[16 + c1];
        float g0v = sW[c0],      g1v = sW[c1];
        a_dy[2 * v]     = v0 ? (half_t)dg0 : (half_t)0.f;
        a_dy[2 * v + 1] = v1 ? (half_t)dg1 : (half_t)0.f;
        a_dx[2 * v]     = v0 ? (half_t)g0v : (half_t)0.f;
        a_dx[2 * v + 1] = v1 ? (half_t)g1v : (half_t)0.f;
    }

    const int colL = tc + m;   // column within the 64-wide tile
    v8f zero = {};

    v16h bHg   = load_B(sHg,   colL, ly0, hi);
    v8f c_dy = __builtin_amdgcn_wmma_f32_16x16x32_f16(false, a_dy, false, bHg,
                                                      (short)0, zero, false, false);
    v16h bHdg  = load_B(sHdg,  colL, ly0, hi);
    v8f c_dx = __builtin_amdgcn_wmma_f32_16x16x32_f16(false, a_dx, false, bHdg,
                                                      (short)0, zero, false, false);
    v16h bAHg  = load_B(sAHg,  colL, ly0, hi);
    v8f c_wdy = __builtin_amdgcn_wmma_f32_16x16x32_f16(false, a_dy, false, bAHg,
                                                       (short)0, zero, false, false);
    v16h bAHdg = load_B(sAHdg, colL, ly0, hi);
    v8f c_wdx = __builtin_amdgcn_wmma_f32_16x16x32_f16(false, a_dx, false, bAHdg,
                                                       (short)0, zero, false, false);

    // ---- Phase 4: scale by EXTGRADFAC and store (C layout: VGPR r -> M=r / r+8) ----
    const size_t planeSz = (size_t)1024 * 1024;
    float* g0p = gimg  + ((size_t)b * 2 + 0) * planeSz;
    float* g1p = gimg  + ((size_t)b * 2 + 1) * planeSz;
    float* w0p = gimgW + ((size_t)b * 2 + 0) * planeSz;
    float* w1p = gimgW + ((size_t)b * 2 + 1) * planeSz;
    const int colg = x0 + tc + m;
#pragma unroll
    for (int r = 0; r < 8; ++r) {
        int rowg = y0 + ly0 + r + hi * 8;
        size_t o = (size_t)rowg * 1024 + colg;
        g0p[o] = c_dy[r]  * 10.f;
        g1p[o] = c_dx[r]  * 10.f;
        w0p[o] = c_wdy[r] * 10.f;
        w1p[o] = c_wdx[r] * 10.f;
    }

    // ---- Phase 5: fused weighted-MSE loss on the interior (covered once) ----
    // pred comes back out of LDS (sIn untouched since phase 1); snake streams
    // in as coalesced float4 loads.
    float lsum = 0.f;
    for (int i = t; i < TILE_H * (TILE_W / 4); i += 256) {   // 512 chunks
        int iy = i >> 4, ic = (i & 15) * 4;
        const float4 sn4 = *(const float4*)&snakeB[(size_t)(y0 + iy) * 1024 + x0 + ic];
        const float4 v4  = *(const float4*)&sIn[(iy + 4) * IN_PITCH + 4 + ic];
        const float sn[4] = { sn4.x, sn4.y, sn4.z, sn4.w };
        const float pv[4] = { v4.x,  v4.y,  v4.z,  v4.w  };
#pragma unroll
        for (int e = 0; e < 4; ++e) {
            float sd = sn[e] > 0.f ? 15.f : sn[e];    // dmax clamp
            sd = sd < 0.f ? sd * 2.f : sd;            // enhancement_factor
            float wgt = sd < 0.f ? 2.5f : 1.f;        // 1 + neg*1.5
            float d = pv[e] - sd;
            lsum += d * d * wgt;
        }
    }

    // ---- Phase 6: deterministic per-block loss partial ----
    sRed[t] = lsum;
    __syncthreads();
#pragma unroll
    for (int s = 128; s > 0; s >>= 1) {
        if (t < s) sRed[t] += sRed[t + s];
        __syncthreads();
    }
    if (t == 0) {
        int blk = (blockIdx.z * gridDim.y + blockIdx.y) * gridDim.x + blockIdx.x;
        partials[blk] = sRed[0];
    }
}

__global__ __launch_bounds__(256)
void snake_loss_finish(const float* __restrict__ partials, int n,
                       float* __restrict__ out)
{
    __shared__ float sRed[256];
    float s = 0.f;
    for (int i = threadIdx.x; i < n; i += 256) s += partials[i];
    sRed[threadIdx.x] = s;
    __syncthreads();
#pragma unroll
    for (int k = 128; k > 0; k >>= 1) {
        if (threadIdx.x < k) sRed[threadIdx.x] += sRed[threadIdx.x + k];
        __syncthreads();
    }
    if (threadIdx.x == 0) out[0] = sRed[0] * (1.0f / 16777216.0f);
}

extern "C" void kernel_launch(void* const* d_in, const int* in_sizes, int n_in,
                              void* d_out, int out_size, void* d_ws, size_t ws_size,
                              hipStream_t stream)
{
    const float* pred  = (const float*)d_in[0];
    const float* snake = (const float*)d_in[1];
    float* out   = (float*)d_out;
    float* gimg  = out + 1;
    float* gimgW = gimg + (size_t)16 * 2 * 1024 * 1024;
    float* partials = (float*)d_ws;

    // Recompute exact filter weights on host (stdev = 2.0, r = 4).
    Filt f;
    {
        float g[9], dg[9], gs = 0.f, ds = 0.f;
        for (int i = 0; i < 9; ++i) {
            float x = (float)(i - 4);
            g[i]  = expf(-x * x / 8.f);
            gs   += g[i];
            dg[i] = (-x / 4.f) * expf(-x * x / 8.f);
            ds   += fabsf(dg[i]);
        }
        for (int i = 0; i < 9; ++i) { f.g[i] = g[i] / gs; f.dg[i] = dg[i] / ds; }
    }

    dim3 grid(1024 / TILE_W, 1024 / TILE_H, 16);   // 16 x 32 x 16 = 8192 blocks
    snake_conv_loss_kernel<<<grid, 256, 0, stream>>>(pred, snake, gimg, gimgW,
                                                     partials, f);
    snake_loss_finish<<<1, 256, 0, stream>>>(partials, 16 * 32 * 16, out);
}
